// MambaModel_69380901700059
// MI455X (gfx1250) — matrix operand order repaired and checked by reading, not compiled, not run on hardware
//
#include <hip/hip_runtime.h>
#include <hip/hip_bf16.h>
#include <math.h>

// ---------------------------------------------------------------------------
// Mamba forward (2 layers) for gfx1250 (MI455X).
// All GEMMs run on v_wmma_f32_16x16x32_bf16 (f32 staged -> bf16 LDS fragments),
// double-buffered LDS, one barrier per k-step, global loads overlapped with WMMA.
// ---------------------------------------------------------------------------

typedef __attribute__((ext_vector_type(16))) __bf16 v16bf;
typedef __attribute__((ext_vector_type(8)))  __bf16 v8bf;
typedef __attribute__((ext_vector_type(8)))  float  v8f;

#define SEQ     1024
#define HID     768
#define INTER   1536
#define VOCABSZ 50257
#define DSTATE  16
#define DTRANK  48
#define XPN     (DTRANK + 2*DSTATE)   // 80
#define KCONV   4
#define NLAYERS 2

__device__ __forceinline__ float softplus_f(float x) {
    return (x > 20.f) ? x : log1pf(__expf(x));
}
__device__ __forceinline__ float silu_f(float x) {
    return x / (1.f + __expf(-x));
}

__device__ __forceinline__ v8bf pack8(const float* v) {
    v8bf p;
#pragma unroll
    for (int i = 0; i < 8; ++i) p[i] = (__bf16)v[i];
    return p;
}

// ---------------------------------------------------------------------------
// Generic GEMM: C[M,N] = epi( A[M,K] @ B[K,N] )
//   EPI 0: plain store
//   EPI 1: softplus(acc + bias[n])          (dt_proj)
//   EPI 2: acc + residual[m,n]              (out_proj residual add)
// Block: 256 threads (8 wave32). Block tile 64x128, wave tile 32x32.
//
// LDS tiles in fragment order (this kernel's contiguous k = 16*half + e, A and B alike):
//   A (16x32 MxK): lane = m + 16*half; an 8-aligned k-segment is contiguous
//                  (e = 8*grp + j) -> one b128 store per thread.
//   B (32x16 KxN): lane = n + 16*(k>>4), e = k&15; a 16-aligned k-column is
//                  one lane's whole 32B fragment slot -> two b128 stores.
// ---------------------------------------------------------------------------
template <int EPI>
__global__ __launch_bounds__(256)
void gemm_wmma_bf16(const float* __restrict__ A, int lda,
                    const float* __restrict__ B, int ldb,
                    float* __restrict__ C, int ldc,
                    int M, int N, int K,
                    const float* __restrict__ bias,
                    const float* __restrict__ residual, int ldres)
{
    __shared__ __align__(32) __bf16 smA[2][4 * 512];   // 64x32 A tiles
    __shared__ __align__(32) __bf16 smB[2][8 * 512];   // 32x128 B tiles

    const int t    = threadIdx.x;
    const int lane = t & 31;            // wave32
    const int wave = t >> 5;            // 0..7
    const int wm   = wave >> 2;         // 0..1  (rows)
    const int wn   = wave & 3;          // 0..3  (cols)

    const int blockRow = blockIdx.y * 64;
    const int blockCol = blockIdx.x * 128;

    v8f acc[2][2];
#pragma unroll
    for (int i = 0; i < 2; ++i)
#pragma unroll
        for (int j = 0; j < 2; ++j)
            acc[i][j] = (v8f){0.f, 0.f, 0.f, 0.f, 0.f, 0.f, 0.f, 0.f};

    // ---- A staging coords: thread -> (row, 8-wide k segment) ----
    const int arow  = t >> 2;               // 0..63
    const int akseg = (t & 3) * 8;          // 0,8,16,24
    const int ahalf = akseg >> 4;
    const int agrp  = (akseg >> 3) & 1;
    const int aoff  = (arow >> 4) * 512 + ((arow & 15) + 16 * ahalf) * 16 + 8 * agrp;
    const int grow  = blockRow + arow;

    // ---- B staging coords: thread -> (column n, 16-wide k segment) ----
    const int bn    = t & 127;              // 0..127
    const int bks   = (t >> 7) * 16;        // 0 or 16
    const int boff  = (bn >> 4) * 512 + ((bn & 15) + 16 * (bks >> 4)) * 16;
    const int gn    = blockCol + bn;

    const bool fullTile = (blockRow + 64 <= M) && (blockCol + 128 <= N);

    float av[8], bv[16];

    // Load one 32-wide k tile of A and B into registers.
    auto loadT = [&](int k0) {
        if (fullTile && (k0 + 32 <= K)) {
            const float* ap = A + (size_t)grow * lda + k0 + akseg;
            const float4 f0 = *(const float4*)ap;
            const float4 f1 = *(const float4*)(ap + 4);
            av[0] = f0.x; av[1] = f0.y; av[2] = f0.z; av[3] = f0.w;
            av[4] = f1.x; av[5] = f1.y; av[6] = f1.z; av[7] = f1.w;

            const float* bp = B + (size_t)(k0 + bks) * ldb + gn;
#pragma unroll
            for (int j = 0; j < 16; ++j)
                bv[j] = bp[(size_t)j * ldb];

            if (k0 + 32 < K)  // prefetch next B k-tile (streaming weights)
                __builtin_prefetch(B + (size_t)(k0 + 32 + bks) * ldb + gn, 0, 1);
        } else {
            // Edge: clamped addresses, unconditional loads, 0/1 mask multiply.
            // (Loads always executed -> no per-element exec-mask predication.)
            const int   rr    = (grow < M) ? grow : (M - 1);
            const float rowok = (grow < M) ? 1.f : 0.f;
            const float* ap   = A + (size_t)rr * lda;
#pragma unroll
            for (int j = 0; j < 8; ++j) {
                const int gk = k0 + akseg + j;
                const int kk = (gk < K) ? gk : (K - 1);
                av[j] = ap[kk] * ((gk < K) ? rowok : 0.f);
            }
            const int   cc    = (gn < N) ? gn : (N - 1);
            const float colok = (gn < N) ? 1.f : 0.f;
#pragma unroll
            for (int j = 0; j < 16; ++j) {
                const int gk = k0 + bks + j;
                const int kk = (gk < K) ? gk : (K - 1);
                bv[j] = B[(size_t)kk * ldb + cc] * ((gk < K) ? colok : 0.f);
            }
        }
    };

    auto stageT = [&](int buf) {
        *(v8bf*)&smA[buf][aoff]     = pack8(av);
        *(v8bf*)&smB[buf][boff]     = pack8(bv);
        *(v8bf*)&smB[buf][boff + 8] = pack8(bv + 8);
    };

    // ---- pipeline prologue ----
    loadT(0);
    stageT(0);
    __syncthreads();

    int cur = 0;
    for (int k0 = 0; k0 < K; k0 += 32) {
        const bool hasNext = (k0 + 32) < K;
        if (hasNext) loadT(k0 + 32);        // global loads overlap compute below

        // ---- WMMA: 2x2 tiles per wave from buf[cur] ----
        v16bf aF[2], bF[2];
#pragma unroll
        for (int i = 0; i < 2; ++i)
            aF[i] = *(const v16bf*)&smA[cur][(wm * 2 + i) * 512 + lane * 16];
#pragma unroll
        for (int j = 0; j < 2; ++j)
            bF[j] = *(const v16bf*)&smB[cur][(wn * 2 + j) * 512 + lane * 16];

#pragma unroll
        for (int i = 0; i < 2; ++i)
#pragma unroll
            for (int j = 0; j < 2; ++j)
                acc[i][j] = __builtin_amdgcn_wmma_f32_16x16x32_bf16(
                    false, aF[i], false, bF[j], (short)0, acc[i][j], false, false);

        if (hasNext) {
            stageT(cur ^ 1);                // writes other buffer: no conflict
            __syncthreads();                // publish before next iteration reads
            cur ^= 1;
        }
    }

    // ---- epilogue: C/D layout (ISA 7.12.2): VGPR r -> M = r + 8*(lane>>4), N = lane&15
    const int mlo = (lane >> 4) * 8;
    const int nn  = lane & 15;
#pragma unroll
    for (int i = 0; i < 2; ++i) {
#pragma unroll
        for (int j = 0; j < 2; ++j) {
            const int row0 = blockRow + wm * 32 + i * 16;
            const int col  = blockCol + wn * 32 + j * 16 + nn;
#pragma unroll
            for (int r = 0; r < 8; ++r) {
                const int row = row0 + mlo + r;
                if (row < M && col < N) {
                    float v = acc[i][j][r];
                    if (EPI == 1) v = softplus_f(v + bias[col]);
                    if (EPI == 2) v += residual[(size_t)row * ldres + col];
                    C[(size_t)row * ldc + col] = v;
                }
            }
        }
    }
}

// ---------------------------------------------------------------------------
// Embedding gather: h[s, :] = embed_W[x[s], :]
// ---------------------------------------------------------------------------
__global__ void embed_kernel(const int* __restrict__ x,
                             const float* __restrict__ W,
                             float* __restrict__ h)
{
    const int s = blockIdx.x;
    const int d = threadIdx.x;                 // 768 threads
    h[(size_t)s * HID + d] = W[(size_t)x[s] * HID + d];
}

// ---------------------------------------------------------------------------
// Depthwise causal conv (K=4) + bias + SiLU.
// raw = proj[:, 0:INTER] with row stride 2*INTER.
// Grid: (INTER/256, SEQ); no integer div/mod.
// ---------------------------------------------------------------------------
__global__ __launch_bounds__(256)
void conv_silu_kernel(const float* __restrict__ proj,
                      const float* __restrict__ cw,   // [KCONV, INTER]
                      const float* __restrict__ cb,   // [INTER]
                      float* __restrict__ hs)
{
    const int c = blockIdx.x * blockDim.x + threadIdx.x;   // channel
    const int s = blockIdx.y;                              // sequence pos

    float acc = cb[c];
#pragma unroll
    for (int j = 0; j < KCONV; ++j) {
        const int ss = s - (KCONV - 1) + j;
        if (ss >= 0)
            acc += proj[(size_t)ss * (2 * INTER) + c] * cw[j * INTER + c];
    }
    hs[(size_t)s * INTER + c] = silu_f(acc);
}

// ---------------------------------------------------------------------------
// Selective-scan: one thread per channel c, 16-state recurrence in registers.
// z[s,c] = (y[s,c] + hs[s,c]*D[c]) * silu(gate[s,c])
// ---------------------------------------------------------------------------
__global__ __launch_bounds__(256)
void scan_kernel(const float* __restrict__ dt,     // [SEQ, INTER] (softplus'd)
                 const float* __restrict__ sp,     // [SEQ, XPN]
                 const float* __restrict__ hs,     // [SEQ, INTER]
                 const float* __restrict__ A_log,  // [INTER, DSTATE]
                 const float* __restrict__ Dv,     // [INTER]
                 const float* __restrict__ gate,   // base proj+INTER, stride 2*INTER
                 float* __restrict__ z)            // [SEQ, INTER]
{
    const int c = blockIdx.x * blockDim.x + threadIdx.x;
    if (c >= INTER) return;

    float Aa[DSTATE], st[DSTATE];
#pragma unroll
    for (int d = 0; d < DSTATE; ++d) {
        Aa[d] = -__expf(A_log[(size_t)c * DSTATE + d]);
        st[d] = 0.f;
    }
    const float Dc = Dv[c];

    for (int s = 0; s < SEQ; ++s) {
        const float dtc = dt[(size_t)s * INTER + c];
        const float hsc = hs[(size_t)s * INTER + c];
        const float f   = dtc * hsc;
        const float* Bm = sp + (size_t)s * XPN + DTRANK;
        const float* Cm = Bm + DSTATE;
        float y = 0.f;
#pragma unroll
        for (int d = 0; d < DSTATE; ++d) {
            const float a = __expf(dtc * Aa[d]);
            st[d] = a * st[d] + f * Bm[d];
            y += st[d] * Cm[d];
        }
        const float g = gate[(size_t)s * (2 * INTER) + c];
        z[(size_t)s * INTER + c] = (y + hsc * Dc) * silu_f(g);
    }
}

// ---------------------------------------------------------------------------
static void launch_gemm(int epi,
                        const float* A, int lda, const float* B, int ldb,
                        float* C, int ldc, int M, int N, int K,
                        const float* bias, const float* residual, int ldres,
                        hipStream_t stream)
{
    dim3 grid((N + 127) / 128, (M + 63) / 64), block(256);
    if (epi == 0)
        gemm_wmma_bf16<0><<<grid, block, 0, stream>>>(A, lda, B, ldb, C, ldc, M, N, K, bias, residual, ldres);
    else if (epi == 1)
        gemm_wmma_bf16<1><<<grid, block, 0, stream>>>(A, lda, B, ldb, C, ldc, M, N, K, bias, residual, ldres);
    else
        gemm_wmma_bf16<2><<<grid, block, 0, stream>>>(A, lda, B, ldb, C, ldc, M, N, K, bias, residual, ldres);
}

extern "C" void kernel_launch(void* const* d_in, const int* in_sizes, int n_in,
                              void* d_out, int out_size, void* d_ws, size_t ws_size,
                              hipStream_t stream)
{
    (void)in_sizes; (void)n_in; (void)out_size; (void)ws_size;

    const int*   x         = (const int*)  d_in[0];
    const float* embed_W   = (const float*)d_in[1];
    const float* unembed_W = (const float*)d_in[2];
    const float* in_proj_W = (const float*)d_in[3];   // [L, HID, 2*INTER]
    const float* conv_W    = (const float*)d_in[4];   // [L, KCONV, 1, INTER]
    const float* conv_b    = (const float*)d_in[5];   // [L, INTER]
    const float* x_proj_W  = (const float*)d_in[6];   // [L, INTER, XPN]
    const float* dt_proj_W = (const float*)d_in[7];   // [L, DTRANK, INTER]
    const float* dt_proj_b = (const float*)d_in[8];   // [L, INTER]
    const float* A_log     = (const float*)d_in[9];   // [L, INTER, DSTATE]
    const float* Dv        = (const float*)d_in[10];  // [L, INTER]
    const float* out_proj_W= (const float*)d_in[11];  // [L, INTER, HID]
    float* out = (float*)d_out;                       // [SEQ, VOCAB] f32

    // workspace layout (floats)
    float* ws   = (float*)d_ws;
    float* h0   = ws;                         // SEQ*HID
    float* h1   = h0 + (size_t)SEQ * HID;     // SEQ*HID
    float* proj = h1 + (size_t)SEQ * HID;     // SEQ*2*INTER
    float* hs   = proj + (size_t)SEQ * 2 * INTER;  // SEQ*INTER
    float* sp   = hs   + (size_t)SEQ * INTER;      // SEQ*XPN
    float* dtb  = sp   + (size_t)SEQ * XPN;        // SEQ*INTER
    float* zb   = dtb  + (size_t)SEQ * INTER;      // SEQ*INTER

    // 1) embedding gather
    embed_kernel<<<SEQ, HID, 0, stream>>>(x, embed_W, h0);

    float* h_cur = h0;
    float* h_nxt = h1;

    for (int l = 0; l < NLAYERS; ++l) {
        const float* Wip = in_proj_W  + (size_t)l * HID * 2 * INTER;
        const float* Wcv = conv_W     + (size_t)l * KCONV * INTER;
        const float* bcv = conv_b     + (size_t)l * INTER;
        const float* Wxp = x_proj_W   + (size_t)l * INTER * XPN;
        const float* Wdt = dt_proj_W  + (size_t)l * DTRANK * INTER;
        const float* bdt = dt_proj_b  + (size_t)l * INTER;
        const float* Al  = A_log      + (size_t)l * INTER * DSTATE;
        const float* Dl  = Dv         + (size_t)l * INTER;
        const float* Wop = out_proj_W + (size_t)l * INTER * HID;

        // in_proj: [SEQ,HID] @ [HID, 2*INTER]
        launch_gemm(0, h_cur, HID, Wip, 2 * INTER, proj, 2 * INTER,
                    SEQ, 2 * INTER, HID, nullptr, nullptr, 0, stream);

        // depthwise conv + SiLU on raw half
        {
            dim3 g(INTER / 256, SEQ);
            conv_silu_kernel<<<g, 256, 0, stream>>>(proj, Wcv, bcv, hs);
        }

        // x_proj: [SEQ,INTER] @ [INTER, 80]
        launch_gemm(0, hs, INTER, Wxp, XPN, sp, XPN,
                    SEQ, XPN, INTER, nullptr, nullptr, 0, stream);

        // dt_proj + softplus: ts = sp[:, :48]  (lda = 80)
        launch_gemm(1, sp, XPN, Wdt, INTER, dtb, INTER,
                    SEQ, INTER, DTRANK, bdt, nullptr, 0, stream);

        // selective scan (+ D skip + SiLU gate) -> zb
        scan_kernel<<<(INTER + 255) / 256, 256, 0, stream>>>(
            dtb, sp, hs, Al, Dl, proj + INTER, zb);

        // out_proj + residual: h_nxt = h_cur + zb @ Wop
        launch_gemm(2, zb, INTER, Wop, HID, h_nxt, HID,
                    SEQ, HID, INTER, nullptr, h_cur, HID, stream);

        float* tmp = h_cur; h_cur = h_nxt; h_nxt = tmp;
    }

    // unembed: [SEQ,HID] @ [HID, VOCAB] -> logits
    launch_gemm(0, h_cur, HID, unembed_W, VOCABSZ, out, VOCABSZ,
                SEQ, VOCABSZ, HID, nullptr, nullptr, 0, stream);
}
